// GNNCritic_89026082111548
// MI455X (gfx1250) — matrix operand order repaired and checked by reading, not compiled, not run on hardware
//
#include <hip/hip_runtime.h>
#include <hip/hip_bf16.h>

typedef __attribute__((ext_vector_type(16))) __bf16 v16bf;
typedef __attribute__((ext_vector_type(8)))  float  v8f;

#define BATCH 8
#define HID   64

// ---------------- degree / normalization ----------------
__global__ void k_init_deg(float* __restrict__ deg, int N) {
  int i = blockIdx.x * blockDim.x + threadIdx.x;
  if (i < N) deg[i] = 1.0f;
}

__global__ void k_deg_accum(const int* __restrict__ dst, float* __restrict__ deg, int E) {
  int i = blockIdx.x * blockDim.x + threadIdx.x;
  if (i < E) atomicAdd(&deg[dst[i]], 1.0f);
}

__global__ void k_dis(const float* __restrict__ deg, float* __restrict__ dis, int N) {
  int i = blockIdx.x * blockDim.x + threadIdx.x;
  if (i < N) dis[i] = rsqrtf(deg[i]);
}

// ---------------- bf16 WMMA GEMM: Y[M,64] = X[M,K] @ W[K,64] ----------------
// One wave computes exactly one 16-row x 64-col strip (exact grid, no strip
// loop -> nothing loop-invariant to hoist/spill). A fragments (KS x 8 VGPRs)
// are loaded once and reused across the 4 column tiles; each column tile keeps
// only one 8-VGPR accumulator plus one transient 8-VGPR B fragment from LDS.
template<int K>
__global__ void __launch_bounds__(256) k_gemm_wmma(const float* __restrict__ X,
                                                   const float* __restrict__ W,
                                                   float* __restrict__ Y, int Mstrips) {
  constexpr int KS = K / 32;
  __shared__ alignas(32) __bf16 Wlds[KS * 4 * 32 * 16];

  int tid = threadIdx.x;
  // Stage W as bf16 B-fragments: Wlds[ks][ct][lane][e] = W[ks*32 + ((lane&16)>>1) + (e&7) + ((e&8)<<1)][ct*16 + (lane&15)]
  for (int idx = tid; idx < K * 64; idx += 256) {
    int e    = idx & 15;
    int lane = (idx >> 4) & 31;
    int ct   = (idx >> 9) & 3;
    int ks   = idx >> 11;
    int k    = ks * 32 + ((lane & 16) >> 1) + (e & 7) + ((e & 8) << 1);
    int col  = ct * 16 + (lane & 15);
    Wlds[idx] = (__bf16)W[k * 64 + col];
  }
  __syncthreads();

  int wid   = tid >> 5;
  int lane  = tid & 31;
  int half8 = (lane & 16) >> 1;  // 0 for lanes 0-15, 8 for lanes 16-31

  int strip = blockIdx.x * 8 + wid;
  if (strip >= Mstrips) return;

  // Load all A fragments for this strip (ISA layout:
  // lane<16 -> K {0..7,16..23}; lane>=16 -> K {8..15,24..31} per 32-slice).
  v16bf a[KS];
  const float* xr = X + (size_t)(strip * 16 + (lane & 15)) * K;
#pragma unroll
  for (int ks = 0; ks < KS; ++ks) {
    const float4* p = reinterpret_cast<const float4*>(xr + ks * 32 + half8);
    float4 f0 = p[0], f1 = p[1], f2 = p[4], f3 = p[5];
    a[ks][0]=(__bf16)f0.x;  a[ks][1]=(__bf16)f0.y;  a[ks][2]=(__bf16)f0.z;  a[ks][3]=(__bf16)f0.w;
    a[ks][4]=(__bf16)f1.x;  a[ks][5]=(__bf16)f1.y;  a[ks][6]=(__bf16)f1.z;  a[ks][7]=(__bf16)f1.w;
    a[ks][8]=(__bf16)f2.x;  a[ks][9]=(__bf16)f2.y;  a[ks][10]=(__bf16)f2.z; a[ks][11]=(__bf16)f2.w;
    a[ks][12]=(__bf16)f3.x; a[ks][13]=(__bf16)f3.y; a[ks][14]=(__bf16)f3.z; a[ks][15]=(__bf16)f3.w;
  }

#pragma unroll
  for (int ct = 0; ct < 4; ++ct) {
    v8f acc = (v8f)(0.0f);
#pragma unroll
    for (int ks = 0; ks < KS; ++ks) {
      const v16bf b =
          *reinterpret_cast<const v16bf*>(&Wlds[(((ks * 4 + ct) * 32) + lane) * 16]);
      acc = __builtin_amdgcn_wmma_f32_16x16x32_bf16(
          false, a[ks], false, b, (short)0, acc, false, false);
    }
    // C/D layout: lane<16 -> rows 0..7, lane>=16 -> rows 8..15; col = ct*16 + (lane&15)
#pragma unroll
    for (int r = 0; r < 8; ++r) {
      Y[(size_t)(strip * 16 + half8 + r) * 64 + ct * 16 + (lane & 15)] = acc[r];
    }
  }
}

// ---------------- edge scatter: agg[:,dst,:] += hw[:,src,:] * enorm ----------------
// One wave per edge (uniform src/dst), lane handles 2 features (32*2 = 64),
// edge reused across all 8 batch images. Atomics land in L2 (41MB << 192MB).
__global__ void k_scatter(const float* __restrict__ hw, float* __restrict__ agg,
                          const int* __restrict__ src, const int* __restrict__ dst,
                          const float* __restrict__ dis, int E, int N) {
  int wid  = threadIdx.x >> 5;
  int lane = threadIdx.x & 31;
  int e = blockIdx.x * 8 + wid;
  if (e >= E) return;
  int s = src[e], d = dst[e];
  float nrm = dis[s] * dis[d];
  size_t soff = (size_t)s * HID + lane * 2;
  size_t doff = (size_t)d * HID + lane * 2;
#pragma unroll
  for (int b = 0; b < BATCH; ++b) {
    size_t base = (size_t)b * N * HID;
    float2 v = *reinterpret_cast<const float2*>(hw + base + soff);
    atomicAdd(agg + base + doff,     v.x * nrm);
    atomicAdd(agg + base + doff + 1, v.y * nrm);
  }
}

// ---------------- layer epilogue: h = relu(agg + hw*selfnorm + b), in place ----------------
__global__ void k_finalize(const float* __restrict__ hw, float* __restrict__ agg_h,
                           const float* __restrict__ dis, const float* __restrict__ bias,
                           int N, long total) {
  long i = (long)blockIdx.x * blockDim.x + threadIdx.x;
  if (i >= total) return;
  int f = (int)(i & (HID - 1));
  int n = (int)((i >> 6) % N);
  float sn = dis[n]; sn *= sn;                 // self-loop norm = 1/deg
  float v = agg_h[i] + hw[i] * sn + bias[f];
  agg_h[i] = v > 0.f ? v : 0.f;
}

// ---------------- layer-2 epilogue fused with mean-pool ----------------
__global__ void k_pool(const float* __restrict__ hw, const float* __restrict__ agg,
                       const float* __restrict__ dis, const float* __restrict__ bias,
                       float* __restrict__ pooled, int N, int chunkRows) {
  __shared__ float red[256];
  int b = blockIdx.y;
  int f = threadIdx.x & 63;
  int r = threadIdx.x >> 6;       // 0..3
  float bf = bias[f];
  float part = 0.f;
  int n0 = blockIdx.x * chunkRows;
  int n1 = n0 + chunkRows; if (n1 > N) n1 = N;
  for (int n = n0 + r; n < n1; n += 4) {
    float sn = dis[n]; sn *= sn;
    size_t idx = ((size_t)b * N + n) * HID + f;
    float v = agg[idx] + hw[idx] * sn + bf;
    part += (v > 0.f ? v : 0.f);
  }
  red[threadIdx.x] = part;
  __syncthreads();
  if (threadIdx.x < 64) {
    float t = red[f] + red[64 + f] + red[128 + f] + red[192 + f];
    atomicAdd(&pooled[b * HID + f], t);
  }
}

// ---------------- final linear head ----------------
__global__ void k_head(const float* __restrict__ pooled, const float* __restrict__ add,
                       const float* __restrict__ fcw, const float* __restrict__ fcb,
                       float* __restrict__ out, int N) {
  int b = threadIdx.x;
  if (b >= BATCH) return;
  float s = 0.f;
  for (int f = 0; f < HID; ++f) s += pooled[b * HID + f] * fcw[f];
  out[b] = s * (1.0f / (float)N) + add[b] * fcw[HID] + fcb[0];
}

extern "C" void kernel_launch(void* const* d_in, const int* in_sizes, int n_in,
                              void* d_out, int out_size, void* d_ws, size_t ws_size,
                              hipStream_t stream) {
  const float* x   = (const float*)d_in[0];
  const float* add = (const float*)d_in[1];
  const int*   ei  = (const int*)d_in[2];
  const float* W1  = (const float*)d_in[3];
  const float* b1  = (const float*)d_in[4];
  const float* W2  = (const float*)d_in[5];
  const float* b2  = (const float*)d_in[6];
  const float* fcw = (const float*)d_in[7];
  const float* fcb = (const float*)d_in[8];
  float* out = (float*)d_out;

  const int IN = 128;
  int E = in_sizes[2] / 2;
  int N = (int)((long)in_sizes[0] / (BATCH * IN));
  const int* src = ei;
  const int* dst = ei + E;

  // workspace carve-up (256B aligned)
  char* ws = (char*)d_ws;
  size_t off = 0;
  auto carve = [&](size_t bytes) -> char* {
    char* p = ws + off;
    off += (bytes + 255) & ~(size_t)255;
    return p;
  };
  float* deg    = (float*)carve((size_t)N * 4);
  float* dis    = (float*)carve((size_t)N * 4);
  float* pooled = (float*)carve(BATCH * HID * 4);
  size_t hwBytes = (size_t)BATCH * N * HID * 4;        // 40.96 MB
  float* bufA = (float*)carve(hwBytes);                // hw (GEMM out)
  float* bufB = (float*)carve(hwBytes);                // agg -> h

  long total = (long)BATCH * N * HID;
  int Mstrips = (int)(((long)BATCH * N) / 16);         // 10000
  int gemmBlocks = (Mstrips + 7) / 8;                  // exact cover, 8 waves/block

  // 1) normalization coefficients
  k_init_deg<<<(N + 255) / 256, 256, 0, stream>>>(deg, N);
  k_deg_accum<<<(E + 255) / 256, 256, 0, stream>>>(dst, deg, E);
  k_dis<<<(N + 255) / 256, 256, 0, stream>>>(deg, dis, N);

  // 2) layer 1
  k_gemm_wmma<128><<<gemmBlocks, 256, 0, stream>>>(x, W1, bufA, Mstrips);
  hipMemsetAsync(bufB, 0, hwBytes, stream);
  k_scatter<<<(E + 7) / 8, 256, 0, stream>>>(bufA, bufB, src, dst, dis, E, N);
  k_finalize<<<(int)((total + 255) / 256), 256, 0, stream>>>(bufA, bufB, dis, b1, N, total);

  // 3) layer 2 (h1 lives in bufB; GEMM writes bufA, then bufB is reused as agg)
  k_gemm_wmma<64><<<gemmBlocks, 256, 0, stream>>>(bufB, W2, bufA, Mstrips);
  hipMemsetAsync(bufB, 0, hwBytes, stream);
  k_scatter<<<(E + 7) / 8, 256, 0, stream>>>(bufA, bufB, src, dst, dis, E, N);

  // 4) fused epilogue + mean pool
  hipMemsetAsync(pooled, 0, BATCH * HID * 4, stream);
  int chunkRows = 512;
  dim3 pg((N + chunkRows - 1) / chunkRows, BATCH);
  k_pool<<<pg, 256, 0, stream>>>(bufA, bufB, dis, b2, pooled, N, chunkRows);

  // 5) head
  k_head<<<1, 32, 0, stream>>>(pooled, add, fcw, fcb, out, N);
}
